// MILoss_3951369912745
// MI455X (gfx1250) — matrix operand order repaired and checked by reading, not compiled, not run on hardware
//
#include <hip/hip_runtime.h>
#include <stdint.h>

typedef __attribute__((ext_vector_type(16))) _Float16 v16h;
typedef __attribute__((ext_vector_type(8)))  float    v8f;

#define NBINS 64
#define NROWS 65                // 64 bins + 1 dump row for branchless clamping
#define LVOX  262144            // 64*64*64 voxels per sample
#define NSAMP 4
#define KC    128               // voxels per LDS tile iteration
#define WSTRIDE 136             // halves per LDS row: 16B-aligned, bank-friendly
#define VOX_PER_BLOCK 512
#define ITERS (VOX_PER_BLOCK / KC)      // 4
#define CHUNKS (LVOX / VOX_PER_BLOCK)   // 512

// |d| <= 2 guaranteed by the 4-bin footprint; outer cubic is 0 at |d| == 2.
__device__ __forceinline__ float cubic_bspline(float d) {
    float t = fabsf(d);
    float inner = 0.66666667f + t * t * (0.5f * t - 1.0f);   // 2/3 - t^2 + t^3/2
    float b = 2.0f - t;
    float outer = b * b * b * (1.0f / 6.0f);
    return (t < 1.0f) ? inner : outer;
}

// ---------------- Kernel 1: per-sample min / 1/(max-min) ----------------
__global__ void __launch_bounds__(256)
mi_minmax_kernel(const float* __restrict__ tgt, const float* __restrict__ src,
                 float* __restrict__ mm) {
    const int n = blockIdx.x;        // sample
    const int which = blockIdx.y;    // 0 = target, 1 = source
    const int tid = threadIdx.x;
    const float* p = (which ? src : tgt) + n * LVOX;

    float mn = 3.4e38f, mx = -3.4e38f;
    const float4* p4 = (const float4*)p;
    for (int i = tid; i < LVOX / 4; i += 256) {
        float4 v = p4[i];
        mn = fminf(mn, fminf(fminf(v.x, v.y), fminf(v.z, v.w)));
        mx = fmaxf(mx, fmaxf(fmaxf(v.x, v.y), fmaxf(v.z, v.w)));
    }
    __shared__ float smn[256], smx[256];
    smn[tid] = mn; smx[tid] = mx;
    __syncthreads();
    for (int s = 128; s > 0; s >>= 1) {
        if (tid < s) {
            smn[tid] = fminf(smn[tid], smn[tid + s]);
            smx[tid] = fmaxf(smx[tid], smx[tid + s]);
        }
        __syncthreads();
    }
    if (tid == 0) {
        mm[n * 4 + which * 2 + 0] = smn[0];
        mm[n * 4 + which * 2 + 1] = 1.0f / (smx[0] - smn[0] + 1e-12f);
    }
}

// ------- Kernel 2: sparse Parzen weights -> LDS f16 tiles -> WMMA GEMM -------
__global__ void __launch_bounds__(512)
mi_hist_kernel(const float* __restrict__ tgt, const float* __restrict__ src,
               const float* __restrict__ mm, float* __restrict__ hist) {
    __shared__ __attribute__((aligned(16))) _Float16 ldsW[2 * NROWS * WSTRIDE];
    _Float16* Wt = ldsW;
    _Float16* Ws = ldsW + NROWS * WSTRIDE;

    const int tid   = threadIdx.x;
    const int n     = blockIdx.y;
    const int chunk = blockIdx.x;

    const float tmin = mm[n * 4 + 0], tinv = mm[n * 4 + 1];
    const float smin = mm[n * 4 + 2], sinv = mm[n * 4 + 3];

    const int  wave    = tid >> 5;
    const int  lane    = tid & 31;
    const int  mt      = wave >> 2;          // 16x16 tile row (bins_t)
    const int  nt      = wave & 3;           // 16x16 tile col (bins_s)
    const int  laneLow = lane & 15;
    const bool hiHalf  = lane >= 16;

    v8f acc = {0.f, 0.f, 0.f, 0.f, 0.f, 0.f, 0.f, 0.f};

    // ---- one-time zero of both f16 weight tiles (b128 stores) ----
    {
        uint4 z; z.x = z.y = z.z = z.w = 0u;
        uint4* lz = (uint4*)ldsW;
        const int nvec = (2 * NROWS * WSTRIDE * 2) / 16;   // 2210
        for (int i = tid; i < nvec; i += 512) lz[i] = z;
    }
    __syncthreads();

    // weight-writer state: per-thread column ownership across iterations
    const int arr = tid >> 7;            // 0: target, 1: source (tid < 256)
    const int col = tid & (KC - 1);
    _Float16* Wmine = arr ? Ws : Wt;
    int jPrev = 0;                       // previous 4-bin footprint start

    for (int it = 0; it < ITERS; ++it) {
        // ---- sparse weight generation: branchless via dump row 64 ----
        if (tid < 2 * KC) {
            // clear previous footprint of this column (exact 4 bins, unconditional)
            #pragma unroll
            for (int jj = 0; jj < 4; ++jj) {
                int j  = jPrev + jj;
                int jc = ((unsigned)j > 63u) ? 64 : j;
                Wmine[jc * WSTRIDE + col] = (_Float16)0.0f;
            }
            const int g = n * LVOX + chunk * VOX_PER_BLOCK + it * KC + col;
            float v = arr ? src[g] : tgt[g];
            float u = (v - (arr ? smin : tmin)) * (arr ? sinv : tinv) * 64.0f;
            int j0 = (int)floorf(u) - 1;
            #pragma unroll
            for (int jj = 0; jj < 4; ++jj) {
                int j  = j0 + jj;
                int jc = ((unsigned)j > 63u) ? 64 : j;
                float w = cubic_bspline((float)j - u);
                Wmine[jc * WSTRIDE + col] = (_Float16)w;
            }
            jPrev = j0;
        }
        __syncthreads();

        // ---- WMMA: hist_tile += Wt_tile * Ws_tile^T over KC in k-steps of 32 ----
        #pragma unroll
        for (int kk = 0; kk < KC / 32; ++kk) {
            union { v16h h; uint4 q[2]; } A, B;
            // A: 16x32 f16. lanes 0-15: K groups {0..7,16..23}; lanes 16-31: {8..15,24..31}
            const int abase = kk * 32 + (hiHalf ? 8 : 0);
            const int arow  = mt * 16 + laneLow;
            A.q[0] = *(const uint4*)(Wt + arow * WSTRIDE + abase);
            A.q[1] = *(const uint4*)(Wt + arow * WSTRIDE + abase + 16);
            // B: 32x16 f16, B[k][j] = Ws[j][k]. lane = N col; K groups match A.
            const int bbase = kk * 32 + (hiHalf ? 8 : 0);
            const int bcol  = nt * 16 + laneLow;
            B.q[0] = *(const uint4*)(Ws + bcol * WSTRIDE + bbase);
            B.q[1] = *(const uint4*)(Ws + bcol * WSTRIDE + bbase + 16);

            acc = __builtin_amdgcn_wmma_f32_16x16x32_f16(
                false, A.h, false, B.h, (short)0, acc, false, false);
        }
        __syncthreads();
    }

    // ---- split-K reduction into global histogram ----
    float* h = hist + n * NBINS * NBINS;
    #pragma unroll
    for (int v = 0; v < 8; ++v) {
        const int row = mt * 16 + v + (hiHalf ? 8 : 0);   // C/D layout: VGPR v -> M=v / v+8
        const int c   = nt * 16 + laneLow;
        unsafeAtomicAdd(&h[row * NBINS + c], acc[v]);
    }
}

// ---------------- Kernel 3: entropies + NMI scalar ----------------
__global__ void __launch_bounds__(256)
mi_nmi_kernel(const float* __restrict__ hist, float* __restrict__ out) {
    __shared__ float hs[NBINS * NBINS];
    __shared__ float red[256];
    __shared__ float pt[NBINS], ps[NBINS];
    __shared__ float stash[3];
    const int tid = threadIdx.x;
    float loss = 0.0f;

    for (int n = 0; n < NSAMP; ++n) {
        const float* h = hist + n * NBINS * NBINS;
        float tsum = 0.0f;
        for (int i = tid; i < NBINS * NBINS; i += 256) {
            float v = h[i];
            hs[i] = v;
            tsum += v;
        }
        red[tid] = tsum;
        __syncthreads();
        for (int s = 128; s > 0; s >>= 1) {
            if (tid < s) red[tid] += red[tid + s];
            __syncthreads();
        }
        const float invT = 1.0f / red[0];
        __syncthreads();

        if (tid < NBINS) {
            float rs = 0.0f, cs = 0.0f;
            for (int j = 0; j < NBINS; ++j) {
                rs += hs[tid * NBINS + j];
                cs += hs[j * NBINS + tid];
            }
            pt[tid] = rs * invT;
            ps[tid] = cs * invT;
        }
        __syncthreads();

        // joint entropy
        float ej = 0.0f;
        for (int i = tid; i < NBINS * NBINS; i += 256) {
            float p = hs[i] * invT;
            ej -= p * logf(p + 1e-12f);
        }
        red[tid] = ej;
        __syncthreads();
        for (int s = 128; s > 0; s >>= 1) {
            if (tid < s) red[tid] += red[tid + s];
            __syncthreads();
        }
        if (tid == 0) stash[2] = red[0];
        __syncthreads();

        // marginal entropy (target)
        float et = 0.0f;
        if (tid < NBINS) { float p = pt[tid]; et = -p * logf(p + 1e-12f); }
        red[tid] = et;
        __syncthreads();
        for (int s = 128; s > 0; s >>= 1) {
            if (tid < s) red[tid] += red[tid + s];
            __syncthreads();
        }
        if (tid == 0) stash[0] = red[0];
        __syncthreads();

        // marginal entropy (source)
        float es = 0.0f;
        if (tid < NBINS) { float p = ps[tid]; es = -p * logf(p + 1e-12f); }
        red[tid] = es;
        __syncthreads();
        for (int s = 128; s > 0; s >>= 1) {
            if (tid < s) red[tid] += red[tid + s];
            __syncthreads();
        }
        if (tid == 0) loss += (stash[0] + red[0]) / stash[2];
        __syncthreads();
    }
    if (tid == 0) out[0] = -loss * (1.0f / NSAMP);
}

extern "C" void kernel_launch(void* const* d_in, const int* in_sizes, int n_in,
                              void* d_out, int out_size, void* d_ws, size_t ws_size,
                              hipStream_t stream) {
    const float* tgt = (const float*)d_in[0];
    const float* src = (const float*)d_in[1];
    float* mm   = (float*)d_ws;                 // [4 samples][tmin,tinv,smin,sinv]
    float* hist = mm + 16;                      // [4][64][64] f32
    (void)in_sizes; (void)n_in; (void)out_size; (void)ws_size;

    hipMemsetAsync(d_ws, 0, (16 + NSAMP * NBINS * NBINS) * sizeof(float), stream);
    mi_minmax_kernel<<<dim3(NSAMP, 2), 256, 0, stream>>>(tgt, src, mm);
    mi_hist_kernel<<<dim3(CHUNKS, NSAMP), 512, 0, stream>>>(tgt, src, mm, hist);
    mi_nmi_kernel<<<1, 256, 0, stream>>>(hist, (float*)d_out);
}
